// MyUGRNN_21964462752204
// MI455X (gfx1250) — matrix operand order repaired and checked by reading, not compiled, not run on hardware
//
#include <hip/hip_runtime.h>
#include <hip/hip_bf16.h>

// UGRNN: B=64, S=2048, I=H=256.
// Phase 1: XC/XU = X@W[:I] + b  (bf16 WMMA, bandwidth-bound streaming GEMM,
//          X tile staged to LDS as bf16 once per block)
// Phase 2: per-batch-row-independent recurrence, 4 workgroups x 16 rows,
//          h kept in LDS (f32 + bf16 mirror), recurrent weights pre-packed
//          bf16 fragment-major (256 KB -> L2-resident).

typedef __attribute__((ext_vector_type(16))) __bf16 v16bf;
typedef __attribute__((ext_vector_type(8)))  float  v8f;

#define S_LEN 2048
#define B_DIM 64
#define H_DIM 256
#define I_DIM 256
#define WPACK_DWORDS 131072  // 2(which) x 2(gate) x 8(kt) x 16(nt) x 32(lane) x 8(j)

union Frag {
  unsigned int u[8];
  uint4 q[2];
  v16bf v;
};

// f32 x2 -> packed bf16 dword (RNE)
__device__ __forceinline__ unsigned int pack2bf(float lo, float hi) {
#if __has_builtin(__builtin_amdgcn_cvt_pk_bf16_f32)
  auto r = __builtin_amdgcn_cvt_pk_bf16_f32(lo, hi);
  return __builtin_bit_cast(unsigned int, r);
#else
  unsigned int a = __float_as_uint(lo);
  unsigned int b = __float_as_uint(hi);
  a += 0x7FFFu + ((a >> 16) & 1u);
  b += 0x7FFFu + ((b >> 16) & 1u);
  return (a >> 16) | (b & 0xFFFF0000u);
#endif
}

// ---------------------------------------------------------------------------
// Kernel 1: pack Wc/Wu (512x256 f32) into bf16 B-fragment-major layout.
// B 32x16 bf16 k order used here (same as A frag): lane L -> N = L%16; VGPR j -> K = (L/16)*16 + 2j, +1.
// Flat: ((((which*2+g)*8+kt)*16+nt)*32+lane)*8+j ; which 0 = x-rows, 1 = h-rows.
// ---------------------------------------------------------------------------
__global__ __launch_bounds__(256) void pack_weights(const float* __restrict__ Wc,
                                                    const float* __restrict__ Wu,
                                                    unsigned int* __restrict__ wpack) {
  int tid = blockIdx.x * blockDim.x + threadIdx.x;  // 0..131071
  int j     = tid & 7;
  int lane  = (tid >> 3) & 31;
  int nt    = (tid >> 8) & 15;
  int kt    = (tid >> 12) & 7;
  int g     = (tid >> 15) & 1;
  int which = (tid >> 16) & 1;
  const float* W = g ? Wu : Wc;
  int col = nt * 16 + (lane & 15);
  int row = which * 256 + kt * 32 + (lane >> 4) * 16 + 2 * j;
  float lo = W[row * H_DIM + col];
  float hi = W[(row + 1) * H_DIM + col];
  wpack[tid] = pack2bf(lo, hi);
}

// ---------------------------------------------------------------------------
// Kernel 2: XC/XU = X @ W{c,u}[:256] + b.  One block = 16 tokens x 256 cols x
// both gates.  8 waves: waves 0-3 gate c (4 n-tiles each), waves 4-7 gate u.
// X tile converted to bf16 in LDS once; A-frag = 2x ds_load_b128.
// Results stored [s][b][h] so the recurrent kernel reads dense slabs.
// ---------------------------------------------------------------------------
__global__ __launch_bounds__(256) void gemm_in(const float* __restrict__ X,
                                               const unsigned int* __restrict__ wpack,
                                               const float* __restrict__ bc,
                                               const float* __restrict__ bu,
                                               float* __restrict__ xc_ws,
                                               float* __restrict__ xu_ws) {
  __shared__ unsigned int Xbf[16 * 128];  // 16 rows x 256 bf16 cols (as dwords)

  const int tid  = threadIdx.x;
  const int lane = tid & 31;
  const int w    = tid >> 5;
  const int g    = w >> 2;
  const int ng   = w & 3;
  const int half = lane >> 4;
  const int nloc = lane & 15;
  const int rtile = blockIdx.x;  // 16 consecutive tokens

  // Stage + convert the 16x256 f32 tile to bf16 in LDS (16 floats / thread).
  {
    int r = tid >> 4;
    int c = (tid & 15) * 16;
    const float4* src = reinterpret_cast<const float4*>(
        X + (size_t)(rtile * 16 + r) * I_DIM + c);
    float4 f0 = src[0], f1 = src[1], f2 = src[2], f3 = src[3];
    uint4* dst = reinterpret_cast<uint4*>(&Xbf[r * 128 + (c >> 1)]);
    dst[0] = make_uint4(pack2bf(f0.x, f0.y), pack2bf(f0.z, f0.w),
                        pack2bf(f1.x, f1.y), pack2bf(f1.z, f1.w));
    dst[1] = make_uint4(pack2bf(f2.x, f2.y), pack2bf(f2.z, f2.w),
                        pack2bf(f3.x, f3.y), pack2bf(f3.z, f3.w));
  }
  __syncthreads();

  v8f acc[4] = {};

  for (int kt = 0; kt < 8; ++kt) {
    // A 16x32 frag: lane row = L%16; cols kt*32+half*16 .. +15 contiguous.
    Frag a;
    const uint4* ar = reinterpret_cast<const uint4*>(
        &Xbf[nloc * 128 + kt * 16 + half * 8]);
    a.q[0] = ar[0];
    a.q[1] = ar[1];
    // Preload all 4 B fragments, then issue 4 independent wmmas.
    Frag b[4];
#pragma unroll
    for (int i = 0; i < 4; ++i) {
      int nt = ng * 4 + i;
      const uint4* bp = reinterpret_cast<const uint4*>(
          wpack + (size_t)(((g * 8 + kt) * 16 + nt) * 32 + lane) * 8);
      b[i].q[0] = bp[0];
      b[i].q[1] = bp[1];
    }
#pragma unroll
    for (int i = 0; i < 4; ++i)
      acc[i] = __builtin_amdgcn_wmma_f32_16x16x32_bf16(
          false, a.v, false, b[i].v, (short)0, acc[i], false, false);
  }

  float* ws = g ? xu_ws : xc_ws;
  const float* bias = g ? bu : bc;
#pragma unroll
  for (int i = 0; i < 4; ++i) {
    int col = (ng * 4 + i) * 16 + nloc;
    float bv = bias[col];
#pragma unroll
    for (int j = 0; j < 8; ++j) {
      int token = rtile * 16 + j + half * 8;  // D row = j + half*8
      int bidx = token >> 11;
      int sidx = token & 2047;
      ws[((size_t)sidx * B_DIM + bidx) * H_DIM + col] = acc[i][j] + bv;
    }
  }
}

// ---------------------------------------------------------------------------
// Kernel 3: recurrence. Each block owns 16 batch rows (independent chains).
// 512 threads = 16 waves: waves 0-7 gate c (2 n-tiles each), 8-15 gate u.
// ---------------------------------------------------------------------------
__global__ __launch_bounds__(512) void recurrent(const float* __restrict__ h0,
                                                 const unsigned int* __restrict__ wpack,
                                                 const float* __restrict__ xc_ws,
                                                 const float* __restrict__ xu_ws,
                                                 float* __restrict__ out) {
  __shared__ float Hf[16 * 256];          // h (f32) for the convex update
  __shared__ unsigned int Hbf[16 * 128];  // h (bf16) mirror for WMMA A-frags
  __shared__ float Cs[16 * 256];
  __shared__ float Us[16 * 256];

  const int tid  = threadIdx.x;
  const int lane = tid & 31;
  const int w    = tid >> 5;  // 0..15
  const int g    = w >> 3;
  const int ng   = w & 7;
  const int half = lane >> 4;
  const int nloc = lane & 15;
  const int b0   = blockIdx.x * 16;

  // init: 8 contiguous elements per thread
  {
    int e0 = tid * 8;
    int m = e0 >> 8, col = e0 & 255;
    const float4* hp = reinterpret_cast<const float4*>(h0 + (b0 + m) * H_DIM + col);
    float4 h0v = hp[0], h1v = hp[1];
    *reinterpret_cast<float4*>(&Hf[e0]) = h0v;
    *reinterpret_cast<float4*>(&Hf[e0 + 4]) = h1v;
    *reinterpret_cast<uint4*>(&Hbf[e0 >> 1]) =
        make_uint4(pack2bf(h0v.x, h0v.y), pack2bf(h0v.z, h0v.w),
                   pack2bf(h1v.x, h1v.y), pack2bf(h1v.z, h1v.w));
  }
  __syncthreads();

  const unsigned int* wh = wpack + 65536;  // which=1 (hidden rows)
  const float* xw = g ? xu_ws : xc_ws;
  float* gbuf = g ? Us : Cs;

  for (int t = 0; t < S_LEN; ++t) {
    v8f acc[2] = {};
#pragma unroll
    for (int kt = 0; kt < 8; ++kt) {
      Frag a;
      const uint4* ar = reinterpret_cast<const uint4*>(
          &Hbf[nloc * 128 + kt * 16 + half * 8]);
      a.q[0] = ar[0];
      a.q[1] = ar[1];
      Frag b[2];
#pragma unroll
      for (int i = 0; i < 2; ++i) {
        int nt = ng * 2 + i;
        const uint4* bp = reinterpret_cast<const uint4*>(
            wh + (size_t)(((g * 8 + kt) * 16 + nt) * 32 + lane) * 8);
        b[i].q[0] = bp[0];
        b[i].q[1] = bp[1];
      }
#pragma unroll
      for (int i = 0; i < 2; ++i)
        acc[i] = __builtin_amdgcn_wmma_f32_16x16x32_bf16(
            false, a.v, false, b[i].v, (short)0, acc[i], false, false);
    }

    // gate nonlinearity -> LDS for cross-gate combine
#pragma unroll
    for (int i = 0; i < 2; ++i) {
      int col = (ng * 2 + i) * 16 + nloc;
#pragma unroll
      for (int j = 0; j < 8; ++j) {
        int m = j + half * 8;
        float xv = xw[((size_t)t * B_DIM + b0 + m) * H_DIM + col];
        float pre = xv + acc[i][j];
        float val = g ? (1.0f / (1.0f + __expf(-pre))) : tanhf(pre);
        gbuf[m * 256 + col] = val;
      }
    }
    __syncthreads();

    // h_new = u*c + (1-u)*h ; vectorized; refresh f32 + bf16 copies + output
    {
      int e0 = tid * 8;
      int m = e0 >> 8, col = e0 & 255;
      float4 c0 = *reinterpret_cast<const float4*>(&Cs[e0]);
      float4 c1 = *reinterpret_cast<const float4*>(&Cs[e0 + 4]);
      float4 u0 = *reinterpret_cast<const float4*>(&Us[e0]);
      float4 u1 = *reinterpret_cast<const float4*>(&Us[e0 + 4]);
      float4 hA = *reinterpret_cast<const float4*>(&Hf[e0]);
      float4 hB = *reinterpret_cast<const float4*>(&Hf[e0 + 4]);
      float4 n0, n1;
      n0.x = u0.x * c0.x + (1.0f - u0.x) * hA.x;
      n0.y = u0.y * c0.y + (1.0f - u0.y) * hA.y;
      n0.z = u0.z * c0.z + (1.0f - u0.z) * hA.z;
      n0.w = u0.w * c0.w + (1.0f - u0.w) * hA.w;
      n1.x = u1.x * c1.x + (1.0f - u1.x) * hB.x;
      n1.y = u1.y * c1.y + (1.0f - u1.y) * hB.y;
      n1.z = u1.z * c1.z + (1.0f - u1.z) * hB.z;
      n1.w = u1.w * c1.w + (1.0f - u1.w) * hB.w;
      *reinterpret_cast<float4*>(&Hf[e0]) = n0;
      *reinterpret_cast<float4*>(&Hf[e0 + 4]) = n1;
      *reinterpret_cast<uint4*>(&Hbf[e0 >> 1]) =
          make_uint4(pack2bf(n0.x, n0.y), pack2bf(n0.z, n0.w),
                     pack2bf(n1.x, n1.y), pack2bf(n1.z, n1.w));
      float* op = out + (size_t)(b0 + m) * S_LEN * H_DIM + (size_t)t * H_DIM + col;
      *reinterpret_cast<float4*>(op) = n0;
      *reinterpret_cast<float4*>(op + 4) = n1;
      if (t == S_LEN - 1) {
        float* hp = out + (size_t)B_DIM * S_LEN * H_DIM + (b0 + m) * H_DIM + col;
        *reinterpret_cast<float4*>(hp) = n0;
        *reinterpret_cast<float4*>(hp + 4) = n1;
      }
    }
    __syncthreads();

    if (t + 1 < S_LEN)  // pull next step's preactivation slab toward the WGP
      __builtin_prefetch(&xw[((size_t)(t + 1) * B_DIM + b0) * H_DIM], 0, 0);
  }
}

extern "C" void kernel_launch(void* const* d_in, const int* in_sizes, int n_in,
                              void* d_out, int out_size, void* d_ws, size_t ws_size,
                              hipStream_t stream) {
  const float* X  = (const float*)d_in[0];   // (64, 2048, 256)
  const float* h0 = (const float*)d_in[1];   // (1, 64, 256)
  const float* Wc = (const float*)d_in[2];   // (512, 256)
  const float* bc = (const float*)d_in[3];   // (256,)
  const float* Wu = (const float*)d_in[4];   // (512, 256)
  const float* bu = (const float*)d_in[5];   // (256,)
  float* out = (float*)d_out;                // (64,2048,256) ++ (1,64,256)

  unsigned int* wpack = (unsigned int*)d_ws;                       // 512 KB
  float* xc_ws = (float*)((char*)d_ws + (size_t)WPACK_DWORDS * 4); // 134 MB
  float* xu_ws = xc_ws + (size_t)B_DIM * S_LEN * H_DIM;            // 134 MB

  pack_weights<<<WPACK_DWORDS / 256, 256, 0, stream>>>(Wc, Wu, wpack);
  gemm_in<<<(B_DIM * S_LEN) / 16, 256, 0, stream>>>(X, wpack, bc, bu, xc_ws, xu_ws);
  recurrent<<<B_DIM / 16, 512, 0, stream>>>(h0, wpack, xc_ws, xu_ws, out);
}
